// transformer_base_66743791780153
// MI455X (gfx1250) — matrix operand order repaired and checked by reading, not compiled, not run on hardware
//
#include <hip/hip_runtime.h>

// ---------------------------------------------------------------------------
// Fully-fused vector-neuron attention block for MI455X (gfx1250, wave32).
// Pass 1: pack all WMMA weights f32 -> f16 in A-fragment lane order (ws).
// Pass 2: one workgroup (256 thr = 8 waves) per (b, n) point; all channel
//         GEMMs on v_wmma_f32_16x16x32_f16, all intermediates in LDS.
//         attn = q - k + pos is folded into the k-GEMM epilogue.
// ---------------------------------------------------------------------------

typedef __attribute__((ext_vector_type(16))) _Float16 v16h;
typedef __attribute__((ext_vector_type(8)))  _Float16 v8h;
typedef __attribute__((ext_vector_type(8)))  float    v8f;

#define B_    2
#define N_    1024
#define S_    32
#define CIN_  64
#define CE_   128
#define CA_   128
#define SITES 96
#define CSTR  136   // LDS stride (halves) for [site][chan<=128] buffers, %8==0
#define ASTR  392   // LDS stride (halves) for attn2 [s][384], %8==0
#define INV_SCALE 0.08838834764831845f  // 1/sqrt(128)

// dynamic LDS layout (byte offsets, all 16B aligned)
#define OFF_A   0                      // 13312 halves
#define OFF_B   26624                  // 13312 halves
#define OFF_C   53248                  // 13312 halves
#define OFF_AB  79872                  // float [32][128]  (s*128 + chan)
#define OFF_Z0  96256                  // float [3][96]
#define OFF_Q   97408                  // float [3][128]  (d*128 + i)
#define OFF_WR  98944                  // float [128]
#define OFF_XYZ 99456                  // float [96]
#define SMEM_BYTES 99840

// packed-weight offsets in workspace (halves): layout per weight is
//   p = ((kstep*128 + m) * 32) + kh*16 + j,  lane fragment = 32B contiguous
#define PK_E1   0        // K=64  ->  8192 halves
#define PK_E2   8192     // K=128 -> 16384
#define PK_K    24576    // K=128 -> 16384
#define PK_V    40960    // K=128 -> 16384
#define PK_F1   57344    // K=384 -> 49152
#define PK_F2   106496   // K=128 -> 16384
#define PK_TOTAL 122880  // halves (245760 bytes)

// A-fragment element j of lane (kh, m) covers k = kh*8 + ((j>=8)?16:0)
//   + (((j>>1)&3)<<1) + (j&1)   (CDNA5 16-bit A 16x32 VGPR striping)
__global__ __launch_bounds__(256)
void pack_weights_kernel(const float* __restrict__ src,
                         _Float16* __restrict__ dst, int K) {
  const int e = blockIdx.x * 256 + threadIdx.x;
  const int total = (K >> 5) << 12;          // (K/32)*128*32
  if (e >= total) return;
  const int j  = e & 15;
  const int kh = (e >> 4) & 1;
  const int m  = (e >> 5) & 127;
  const int ks = e >> 12;
  const int k  = (ks << 5) + (kh << 3) + ((j >= 8) ? 16 : 0)
               + (((j >> 1) & 3) << 1) + (j & 1);
  dst[e] = (_Float16)src[(size_t)m * K + k];
}

// D = W[128xK] * src[K x 16*NT]; wave w owns output rows [16w,16w+16).
// ATTN epilogue: out = q[d*128+m] - acc + wr[m]*xyz[col]  (attn fold)
template<int K, int NT, bool BIAS, bool RELU, bool ADDPOS, bool ATTN, bool F32OUT>
__device__ __forceinline__ void gemm_stage(
    const _Float16* __restrict__ Wp,          // packed f16 fragments (global)
    const _Float16* __restrict__ src, int sstride,
    _Float16* __restrict__ dsth, int dstride,
    float* __restrict__ dstf,
    const float* __restrict__ bias,
    const float* __restrict__ wr, const float* __restrict__ xyz,
    const float* __restrict__ qb)
{
  const int tid  = threadIdx.x;
  const int wave = tid >> 5;
  const int lane = tid & 31;
  const int m0   = wave << 4;
  const int kh   = lane >> 4;
  const int lidx = lane & 15;

  const _Float16* wrow = Wp + ((size_t)(m0 + lidx) << 5) + (kh << 4);

#pragma unroll
  for (int t = 0; t < NT; ++t) {
    const int c0 = t << 4;
    v8f acc = {0.f, 0.f, 0.f, 0.f, 0.f, 0.f, 0.f, 0.f};
    const _Float16* brow = src + (size_t)(c0 + lidx) * sstride + (kh << 3);
#pragma unroll
    for (int kk = 0; kk < K; kk += 32) {
      // A: 32 contiguous bytes per lane, pre-shuffled -> 2x global b128
      const v8h* ap = (const v8h*)(wrow + ((kk >> 5) << 12));
      const v8h a0 = ap[0];
      const v8h a1 = ap[1];
      v16h afrag;
#pragma unroll
      for (int j = 0; j < 8; ++j) { afrag[j] = a0[j]; afrag[j + 8] = a1[j]; }

      // B: two 16B LDS loads of consecutive k for this lane's column
      const v8h* bp = (const v8h*)(brow + kk);
      const v8h b0 = bp[0];
      const v8h b1 = bp[2];
      v16h bfrag;
#pragma unroll
      for (int j = 0; j < 8; ++j) { bfrag[j] = b0[j]; bfrag[j + 8] = b1[j]; }

      acc = __builtin_amdgcn_wmma_f32_16x16x32_f16(
          false, afrag, false, bfrag, (short)0, acc, false, false);
    }
    // epilogue: rows m0 + r + 8*kh (r=0..7) are contiguous channels in the
    // [site][chan] layout -> pack to one 16B store.
    const int col = c0 + lidx;
    if constexpr (F32OUT) {
      float vv[8];
#pragma unroll
      for (int r = 0; r < 8; ++r) {
        float v = acc[r];
        if constexpr (BIAS) v += bias[m0 + (kh << 3) + r];
        vv[r] = v;
      }
      float* dp = dstf + (size_t)col * 128 + m0 + (kh << 3);
      *(float4*)(dp)     = make_float4(vv[0], vv[1], vv[2], vv[3]);
      *(float4*)(dp + 4) = make_float4(vv[4], vv[5], vv[6], vv[7]);
    } else {
      v8h hv;
#pragma unroll
      for (int r = 0; r < 8; ++r) {
        const int m = m0 + (kh << 3) + r;
        float v = acc[r];
        if constexpr (BIAS)   v += bias[m];
        if constexpr (RELU)   v = fmaxf(v, 0.0f);
        if constexpr (ADDPOS) v += wr[m] * xyz[col];             // v + pos
        if constexpr (ATTN)                                       // q - k + pos
          v = qb[(col >> 5) * 128 + m] - v + wr[m] * xyz[col];
        hv[r] = (_Float16)v;
      }
      *(v8h*)(dsth + (size_t)col * dstride + m0 + (kh << 3)) = hv;
    }
  }
}

__global__ __launch_bounds__(256)
void fused_vn_attn_kernel(
    const float* __restrict__ gxyz, const float* __restrict__ gfts,
    const float* __restrict__ qxyz,
    const _Float16* __restrict__ wpack,
    const float* __restrict__ Wq,  const float* __restrict__ Wr1,
    const float* __restrict__ Wr2, const float* __restrict__ Wstd,
    const float* __restrict__ bf1, const float* __restrict__ bf2,
    float* __restrict__ out)
{
  extern __shared__ char smem[];
  _Float16* sA  = (_Float16*)(smem + OFF_A);
  _Float16* sB  = (_Float16*)(smem + OFF_B);
  _Float16* sC  = (_Float16*)(smem + OFF_C);
  float* s_ab   = (float*)(smem + OFF_AB);
  float* s_z0   = (float*)(smem + OFF_Z0);
  float* s_q    = (float*)(smem + OFF_Q);
  float* s_wr   = (float*)(smem + OFF_WR);
  float* s_xyz  = (float*)(smem + OFF_XYZ);

  const int tid = threadIdx.x;
  const int n   = blockIdx.x & (N_ - 1);
  const int b   = blockIdx.x >> 10;

  // ---- init: wr = W_r2 @ W_r1 (rank-1 pos_rel), xyz_rel, X tile load ----
  if (tid < 128) {
    float acc = 0.f;
#pragma unroll 4
    for (int m = 0; m < 64; ++m) acc += Wr2[tid * 64 + m] * Wr1[m];
    s_wr[tid] = acc;
  }
  if (tid < SITES) {
    const int d = tid >> 5, s = tid & 31;
    s_xyz[tid] = gxyz[(((size_t)b * N_ + n) * S_ + s) * 3 + d]
               - qxyz[((size_t)b * N_ + n) * 3 + d];
  }
  for (int e = tid; e < CIN_ * SITES; e += 256) {
    const int c = e / SITES, site = e % SITES;
    sA[(size_t)site * CSTR + c] = (_Float16)
        gfts[(((size_t)b * CIN_ + c) * 3 + (site >> 5)) * (N_ * S_)
             + (size_t)n * S_ + (site & 31)];
  }
  __syncthreads();

  // ---- e1: emb1 = We1 @ X -> sB ----
  gemm_stage<CIN_, 6, false, false, false, false, false>(wpack + PK_E1,
      sA, CSTR, sB, CSTR, nullptr, nullptr, nullptr, nullptr, nullptr);
  __syncthreads();
  // ---- e2: emb = We2 @ emb1 -> sC ----
  gemm_stage<CE_, 6, false, false, false, false, false>(wpack + PK_E2,
      sB, CSTR, sC, CSTR, nullptr, nullptr, nullptr, nullptr, nullptr);
  __syncthreads();

  // ---- q = Wq @ emb[:, s=0 columns]  (3 sites, VALU) ----
  for (int t = tid; t < 384; t += 256) {
    const int i = t & 127, d = t >> 7;
    const v8h* er = (const v8h*)(sC + (size_t)(d * 32) * CSTR); // site = d*32
    const float* wq = Wq + (size_t)i * CE_;
    float acc = 0.f;
#pragma unroll
    for (int c8 = 0; c8 < 16; ++c8) {
      const v8h hv = er[c8];
#pragma unroll
      for (int j = 0; j < 8; ++j) acc += wq[c8 * 8 + j] * (float)hv[j];
    }
    s_q[d * 128 + i] = acc;
  }
  __syncthreads();

  // ---- k GEMM with fused attn epilogue: sA = q - k + wr*xyz ----
  gemm_stage<CE_, 6, false, false, false, true,  false>(wpack + PK_K,
      sC, CSTR, sA, CSTR, nullptr, nullptr, s_wr, s_xyz, s_q);
  // ---- v GEMM with fused pos add: sB = v + wr*xyz ----
  gemm_stage<CE_, 6, false, false, true,  false, false>(wpack + PK_V,
      sC, CSTR, sB, CSTR, nullptr, nullptr, s_wr, s_xyz, nullptr);
  __syncthreads();

  // ---- z0 = Wstd @ attn  (3 x 96) ----
  for (int t = tid; t < 3 * SITES; t += 256) {
    const int o = t / SITES, site = t % SITES;
    const v8h* ar = (const v8h*)(sA + (size_t)site * CSTR);
    const float* ws = Wstd + o * CA_;
    float acc = 0.f;
#pragma unroll
    for (int c8 = 0; c8 < 16; ++c8) {
      const v8h hv = ar[c8];
#pragma unroll
      for (int j = 0; j < 8; ++j) acc += ws[c8 * 8 + j] * (float)hv[j];
    }
    s_z0[o * SITES + site] = acc;
  }
  __syncthreads();

  // ---- attn2[s][i*3+kd] = sum_j attn[i][j*32+s] * z0[j][kd*32+s] -> sC ----
  for (int e = tid; e < 384 * S_; e += 256) {
    const int s = e / 384, c2 = e % 384;
    const int i = c2 / 3, kd = c2 % 3;
    float acc = 0.f;
#pragma unroll
    for (int j = 0; j < 3; ++j)
      acc += (float)sA[(size_t)(j * 32 + s) * CSTR + i] * s_z0[j * SITES + kd * 32 + s];
    sC[(size_t)s * ASTR + c2] = (_Float16)acc;
  }
  __syncthreads();

  // ---- f1: h = relu(Wf1 @ attn2 + bf1) -> sA ([s<32][chan]) ----
  gemm_stage<384, 2, true, true, false, false, false>(wpack + PK_F1,
      sC, ASTR, sA, CSTR, nullptr, bf1, nullptr, nullptr, nullptr);
  __syncthreads();
  // ---- f2: a = Wf2 @ h + bf2 -> s_ab (f32 [s][128]) ----
  gemm_stage<CA_, 2, true, false, false, false, true>(wpack + PK_F2,
      sA, CSTR, nullptr, 0, s_ab, bf2, nullptr, nullptr, nullptr);
  __syncthreads();

  // ---- softmax over the 32 neighbors per channel ----
  if (tid < 128) {
    float mx = -3.4e38f;
#pragma unroll 4
    for (int s = 0; s < 32; ++s) mx = fmaxf(mx, s_ab[s * 128 + tid]);
    float sum = 0.f;
#pragma unroll 4
    for (int s = 0; s < 32; ++s) {
      const float p = __expf((s_ab[s * 128 + tid] - mx) * INV_SCALE);
      s_ab[s * 128 + tid] = p;
      sum += p;
    }
    const float inv = 1.f / sum;
#pragma unroll 4
    for (int s = 0; s < 32; ++s) s_ab[s * 128 + tid] *= inv;
  }
  __syncthreads();

  // ---- resi[i][d] = sum_s a[s][i] * vpr[i][d*32+s] ----
  for (int t = tid; t < 384; t += 256) {
    const int i = t / 3, d = t % 3;
    float acc = 0.f;
#pragma unroll 4
    for (int s = 0; s < 32; ++s)
      acc += s_ab[s * 128 + i] * (float)sB[(size_t)(d * 32 + s) * CSTR + i];
    out[(((size_t)b * CA_ + i) * 3 + d) * N_ + n] = acc;
  }
}

extern "C" void kernel_launch(void* const* d_in, const int* in_sizes, int n_in,
                              void* d_out, int out_size, void* d_ws, size_t ws_size,
                              hipStream_t stream) {
  (void)in_sizes; (void)n_in; (void)out_size; (void)ws_size;
  const float* gxyz = (const float*)d_in[0];
  const float* gfts = (const float*)d_in[1];
  const float* qxyz = (const float*)d_in[2];
  const float* We1  = (const float*)d_in[3];
  const float* We2  = (const float*)d_in[4];
  const float* Wq   = (const float*)d_in[5];
  const float* Wk   = (const float*)d_in[6];
  const float* Wv   = (const float*)d_in[7];
  const float* Wr1  = (const float*)d_in[8];
  const float* Wr2  = (const float*)d_in[9];
  const float* Wstd = (const float*)d_in[10];
  const float* Wf1  = (const float*)d_in[11];
  const float* bf1  = (const float*)d_in[12];
  const float* Wf2  = (const float*)d_in[13];
  const float* bf2  = (const float*)d_in[14];
  float* out = (float*)d_out;
  _Float16* wpack = (_Float16*)d_ws;

  // Pass 1: pack WMMA weights into f16 fragment order (L2-resident, tiny)
  pack_weights_kernel<<<dim3(( 64 >> 5) << 4), dim3(256), 0, stream>>>(We1, wpack + PK_E1,  64);
  pack_weights_kernel<<<dim3((128 >> 5) << 4), dim3(256), 0, stream>>>(We2, wpack + PK_E2, 128);
  pack_weights_kernel<<<dim3((128 >> 5) << 4), dim3(256), 0, stream>>>(Wk,  wpack + PK_K,  128);
  pack_weights_kernel<<<dim3((128 >> 5) << 4), dim3(256), 0, stream>>>(Wv,  wpack + PK_V,  128);
  pack_weights_kernel<<<dim3((384 >> 5) << 4), dim3(256), 0, stream>>>(Wf1, wpack + PK_F1, 384);
  pack_weights_kernel<<<dim3((128 >> 5) << 4), dim3(256), 0, stream>>>(Wf2, wpack + PK_F2, 128);

  // allow ~100KB dynamic LDS (idempotent, deterministic)
  (void)hipFuncSetAttribute((const void*)fused_vn_attn_kernel,
                            hipFuncAttributeMaxDynamicSharedMemorySize,
                            SMEM_BYTES);

  // Pass 2: fused kernel, one workgroup per (b, n)
  fused_vn_attn_kernel<<<dim3(B_ * N_), dim3(256), SMEM_BYTES, stream>>>(
      gxyz, gfts, qxyz, wpack, Wq, Wr1, Wr2, Wstd, bf1, bf2, out);
}
